// VGG_SMALL_1W1A_72894184948232
// MI455X (gfx1250) — 8-bit matrix operand order repaired and checked by reading, compiled, not run on hardware
//
#include <hip/hip_runtime.h>
#include <hip/hip_bf16.h>


typedef __attribute__((ext_vector_type(8))) int v8i;

__device__ __forceinline__ signed char sgn8(float v) {
    return (v > 0.f) ? (signed char)1 : ((v < 0.f) ? (signed char)-1 : (signed char)0);
}

__global__ __launch_bounds__(128)
void conv0_bn_kernel(const float* __restrict__ x, const float* __restrict__ w,
                     const float* __restrict__ bn, float* __restrict__ out)
{
    const int o   = threadIdx.x;
    const int pix = blockIdx.x;
    const int ox  = pix & 31;
    const int oy  = (pix >> 5) & 31;
    const int n   = pix >> 10;

    float acc = 0.f;
    #pragma unroll
    for (int c = 0; c < 3; ++c) {
        const float* xc = x + (long long)(n * 3 + c) * 1024;
        const float* wc = w + (o * 3 + c) * 9;
        #pragma unroll
        for (int ky = 0; ky < 3; ++ky) {
            const int iy = oy + ky - 1;
            if (iy < 0 || iy >= 32) continue;
            #pragma unroll
            for (int kx = 0; kx < 3; ++kx) {
                const int ix = ox + kx - 1;
                if (ix < 0 || ix >= 32) continue;
                acc += xc[iy * 32 + ix] * wc[ky * 3 + kx];
            }
        }
    }
    const float g = bn[o], b = bn[128 + o], m = bn[256 + o], v = bn[384 + o];
    const float inv = g * rsqrtf(v + 1e-5f);
    out[(long long)pix * 128 + o] = acc * inv + (b - m * inv);
}

__global__ __launch_bounds__(256)
void prep_w_kernel(const float* __restrict__ w, signed char* __restrict__ wT,
                   float* __restrict__ alpha, int Cin, int Cout)
{
    __shared__ float red[256];
    const int o = blockIdx.x, tid = threadIdx.x;
    const int K = Cin * 9;
    const float* wo = w + (long long)o * K;

    float s = 0.f;
    for (int k = tid; k < K; k += 256) s += fabsf(wo[k]);
    red[tid] = s;
    __syncthreads();
    for (int st = 128; st > 0; st >>= 1) {
        if (tid < st) red[tid] += red[tid + st];
        __syncthreads();
    }
    if (tid == 0) alpha[o] = red[0] / (float)K;

    for (int k = tid; k < K; k += 256) {
        const int c = k / 9, r = k - c * 9;
        wT[((long long)r * Cout + o) * Cin + c] = sgn8(wo[k]);
    }
}

__global__ __launch_bounds__(256)
void binarize_kernel(const float4* __restrict__ x, const float* __restrict__ sh,
                     char4* __restrict__ a0, char4* __restrict__ a1,
                     int total4, int C)
{
    const int i = blockIdx.x * 256 + threadIdx.x;
    if (i >= total4) return;
    const int c = (i << 2) & (C - 1);
    const float4 v = x[i];
    char4 r0, r1;
    r0.x = sgn8(v.x + sh[c + 0]);     r1.x = sgn8(v.x + sh[C + c + 0]);
    r0.y = sgn8(v.y + sh[c + 1]);     r1.y = sgn8(v.y + sh[C + c + 1]);
    r0.z = sgn8(v.z + sh[c + 2]);     r1.z = sgn8(v.z + sh[C + c + 2]);
    r0.w = sgn8(v.w + sh[c + 3]);     r1.w = sgn8(v.w + sh[C + c + 3]);
    a0[i] = r0;
    a1[i] = r1;
}

__global__ __launch_bounds__(256)
void bn_ht_kernel(float* __restrict__ y, const float* __restrict__ bn,
                  int total, int C)
{
    const int i = blockIdx.x * 256 + threadIdx.x;
    if (i >= total) return;
    const int c = i & (C - 1);
    const float g = bn[c], b = bn[C + c], m = bn[2 * C + c], v = bn[3 * C + c];
    const float inv = g * rsqrtf(v + 1e-5f);
    const float r = y[i] * inv + (b - m * inv);
    y[i] = fminf(1.f, fmaxf(-1.f, r));
}

__global__ __launch_bounds__(256)
void pool_kernel(const float* __restrict__ in, float* __restrict__ out,
                 int total, int Hout, int Wout, int C)
{
    const int i = blockIdx.x * 256 + threadIdx.x;
    if (i >= total) return;
    int idx = i;
    const int c = idx % C;    idx /= C;
    const int x = idx % Wout; idx /= Wout;
    const int y = idx % Hout;
    const int n = idx / Hout;
    const int Win = 2 * Wout;
    const long long base = (((long long)(n * 2 * Hout + 2 * y)) * Win + 2 * x) * C + c;
    const long long rs = (long long)Win * C;
    float m0 = fmaxf(in[base],      in[base + C]);
    float m1 = fmaxf(in[base + rs], in[base + rs + C]);
    out[i] = fmaxf(m0, m1);
}

__global__ __launch_bounds__(256)
void bconv_wmma_kernel(const signed char* __restrict__ actA,
                       const signed char* __restrict__ actB,
                       const signed char* __restrict__ wT,
                       const float* __restrict__ alpha,
                       const float* __restrict__ sc,
                       float* __restrict__ out,
                       int H, int W, int Cin, int Cout)
{
    __shared__ signed char Btile[32 * 64];

    const int tid  = threadIdx.x;
    const int wave = tid >> 5;
    const int lane = tid & 31;
    const int half = lane >> 4;
    const int l15  = lane & 15;

    const int mtile = blockIdx.x * 128 + wave * 16;
    const int o0    = blockIdx.y * 32;

    const int m   = mtile + l15;
    const int hw  = H * W;
    const int img = m / hw;
    const int rem = m - img * hw;
    const int oy  = rem / W;
    const int ox  = rem - oy * W;

    int akd[8], bkd[8];
    #pragma unroll
    for (int v = 0; v < 8; ++v) {
        akd[v] = (((v & 1) << 2) + (((v >> 1) & 1) << 4) + (((v >> 2) & 1) << 5) + (half << 3)) >> 2;
        bkd[v] = (((v & 1) << 2) + (((v >> 1) & 1) << 4) + (((v >> 2) & 1) << 5) + (half << 3)) >> 2;
    }

    v8i acc1a = {}, acc2a = {};
    v8i acc1b = {}, acc2b = {};

    for (int r = 0; r < 9; ++r) {
        const int dy = r / 3 - 1, dx = r % 3 - 1;
        const int iy = oy + dy, ix = ox + dx;
        const bool inb = (iy >= 0) && (iy < H) && (ix >= 0) && (ix < W);
        const long long abase = ((long long)(img * H + iy) * W + ix) * Cin;
        const signed char* wr = wT + (long long)r * Cout * Cin;

        for (int c0 = 0; c0 < Cin; c0 += 64) {
            __syncthreads();
            {
                const int bn_ = tid >> 3;
                const int kq  = (tid & 7) << 3;
                const int2* src = reinterpret_cast<const int2*>(
                    wr + (long long)(o0 + bn_) * Cin + c0 + kq);
                *reinterpret_cast<int2*>(&Btile[bn_ * 64 + kq]) = *src;
            }
            __syncthreads();

            v8i bfrag0, bfrag1;
            const int* bt0 = reinterpret_cast<const int*>(&Btile[l15 * 64]);
            const int* bt1 = reinterpret_cast<const int*>(&Btile[(16 + l15) * 64]);
            #pragma unroll
            for (int v = 0; v < 8; ++v) { bfrag0[v] = bt0[bkd[v]]; bfrag1[v] = bt1[bkd[v]]; }

            v8i afrag1 = {}, afrag2 = {};
            if (inb) {
                const int* a1p = reinterpret_cast<const int*>(actA + abase + c0);
                const int* a2p = reinterpret_cast<const int*>(actB + abase + c0);
                #pragma unroll
                for (int v = 0; v < 8; ++v) {
                    afrag1[v] = a1p[akd[v]];
                    afrag2[v] = a2p[akd[v]];
                }
            }

            acc1a = __builtin_amdgcn_wmma_i32_16x16x64_iu8(true, afrag1, true, bfrag0,
                                                           acc1a, false, false);
            acc2a = __builtin_amdgcn_wmma_i32_16x16x64_iu8(true, afrag2, true, bfrag0,
                                                           acc2a, false, false);
            acc1b = __builtin_amdgcn_wmma_i32_16x16x64_iu8(true, afrag1, true, bfrag1,
                                                           acc1b, false, false);
            acc2b = __builtin_amdgcn_wmma_i32_16x16x64_iu8(true, afrag2, true, bfrag1,
                                                           acc2b, false, false);
        }
    }

    const float s   = sc[0];
    const float al0 = alpha[o0 + l15];
    const float al1 = alpha[o0 + 16 + l15];
    #pragma unroll
    for (int j = 0; j < 8; ++j) {
        const int row = mtile + (half << 3) + j;
        float* orow = out + (long long)row * Cout + o0 + l15;
        orow[0]  = al0 * ((float)acc1a[j] + s * (float)acc2a[j]);
        orow[16] = al1 * ((float)acc1b[j] + s * (float)acc2b[j]);
    }
}

__global__ __launch_bounds__(256)
void fc_kernel(const float* __restrict__ h, const float* __restrict__ fcw,
               const float* __restrict__ fcb, float* __restrict__ out)
{
    __shared__ float red[256];
    const int n = blockIdx.x, tid = threadIdx.x;
    float acc[10];
    #pragma unroll
    for (int j = 0; j < 10; ++j) acc[j] = 0.f;

    for (int kk = tid; kk < 8192; kk += 256) {
        const int c = kk >> 4, yx = kk & 15;
        const float hv = h[((long long)n * 16 + yx) * 512 + c];
        #pragma unroll
        for (int j = 0; j < 10; ++j) acc[j] += hv * fcw[j * 8192 + kk];
    }
    for (int j = 0; j < 10; ++j) {
        red[tid] = acc[j];
        __syncthreads();
        for (int st = 128; st > 0; st >>= 1) {
            if (tid < st) red[tid] += red[tid + st];
            __syncthreads();
        }
        if (tid == 0) out[n * 10 + j] = red[0] + fcb[j];
        __syncthreads();
    }
}

extern "C" void kernel_launch(void* const* d_in, const int* in_sizes, int n_in,
                              void* d_out, int out_size, void* d_ws, size_t ws_size,
                              hipStream_t stream)
{
    const float* x       = (const float*)d_in[0];
    const float* conv0_w = (const float*)d_in[1];
    const float* bn[6]   = { (const float*)d_in[2], (const float*)d_in[3],
                             (const float*)d_in[4], (const float*)d_in[5],
                             (const float*)d_in[6], (const float*)d_in[7] };
    const float* sh[5]   = { (const float*)d_in[8],  (const float*)d_in[11],
                             (const float*)d_in[14], (const float*)d_in[17],
                             (const float*)d_in[20] };
    const float* w[5]    = { (const float*)d_in[9],  (const float*)d_in[12],
                             (const float*)d_in[15], (const float*)d_in[18],
                             (const float*)d_in[21] };
    const float* sc[5]   = { (const float*)d_in[10], (const float*)d_in[13],
                             (const float*)d_in[16], (const float*)d_in[19],
                             (const float*)d_in[22] };
    const float* fc_w = (const float*)d_in[23];
    const float* fc_b = (const float*)d_in[24];
    float* out = (float*)d_out;

    char* ws = (char*)d_ws;
    float*       F0 = (float*)(ws);
    float*       F1 = (float*)(ws + 134217728LL);
    signed char* A0 = (signed char*)(ws + 268435456LL);
    signed char* A1 = A0 + 33554432LL;
    signed char* WT = A1 + 33554432LL;
    float*       AL = (float*)(WT + 9LL * 512 * 512);

    conv0_bn_kernel<<<256 * 32 * 32, 128, 0, stream>>>(x, conv0_w, bn[0], F0);

    auto binconv = [&](const float* fin, float* fout, int li,
                       int Hh, int Ww, int Cin, int Cout) {
        const int Npix = 256 * Hh * Ww;
        prep_w_kernel<<<Cout, 256, 0, stream>>>(w[li], WT, AL, Cin, Cout);
        const int total4 = (Npix * Cin) >> 2;
        binarize_kernel<<<(total4 + 255) / 256, 256, 0, stream>>>(
            (const float4*)fin, sh[li], (char4*)A0, (char4*)A1, total4, Cin);
        dim3 g(Npix / 128, Cout / 32);
        bconv_wmma_kernel<<<g, 256, 0, stream>>>(A0, A1, WT, AL, sc[li], fout,
                                                 Hh, Ww, Cin, Cout);
    };

    binconv(F0, F1, 0, 32, 32, 128, 128);
    pool_kernel<<<(256 * 16 * 16 * 128 + 255) / 256, 256, 0, stream>>>(
        F1, F0, 256 * 16 * 16 * 128, 16, 16, 128);
    bn_ht_kernel<<<(256 * 16 * 16 * 128 + 255) / 256, 256, 0, stream>>>(
        F0, bn[1], 256 * 16 * 16 * 128, 128);

    binconv(F0, F1, 1, 16, 16, 128, 256);
    bn_ht_kernel<<<(256 * 16 * 16 * 256 + 255) / 256, 256, 0, stream>>>(
        F1, bn[2], 256 * 16 * 16 * 256, 256);

    binconv(F1, F0, 2, 16, 16, 256, 256);
    pool_kernel<<<(256 * 8 * 8 * 256 + 255) / 256, 256, 0, stream>>>(
        F0, F1, 256 * 8 * 8 * 256, 8, 8, 256);
    bn_ht_kernel<<<(256 * 8 * 8 * 256 + 255) / 256, 256, 0, stream>>>(
        F1, bn[3], 256 * 8 * 8 * 256, 256);

    binconv(F1, F0, 3, 8, 8, 256, 512);
    bn_ht_kernel<<<(256 * 8 * 8 * 512 + 255) / 256, 256, 0, stream>>>(
        F0, bn[4], 256 * 8 * 8 * 512, 512);

    binconv(F0, F1, 4, 8, 8, 512, 512);
    pool_kernel<<<(256 * 4 * 4 * 512 + 255) / 256, 256, 0, stream>>>(
        F1, F0, 256 * 4 * 4 * 512, 4, 4, 512);
    bn_ht_kernel<<<(256 * 4 * 4 * 512 + 255) / 256, 256, 0, stream>>>(
        F0, bn[5], 256 * 4 * 4 * 512, 512);

    fc_kernel<<<256, 256, 0, stream>>>(F0, fc_w, fc_b, out);
}
